// GptOssDecoderLayer_89953795048052
// MI455X (gfx1250) — compile-verified, matrix operand order checked by reading, not run on hardware
//
#include <hip/hip_runtime.h>
#include <hip/hip_bf16.h>

// ---------------- problem constants ----------------
#define B_    2
#define S_    1024
#define H_    2880
#define NH_   64
#define NKV_  8
#define HD_   64
#define E_    8
#define F_    2880
#define T_    (B_ * S_)                 // 2048 tokens
#define QKV_N ((NH_ + 2 * NKV_) * HD_)  // 5120
#define QDIM  (NH_ * HD_)               // 4096
#define KVDIM (NKV_ * HD_)              // 512
#define ALPHA_ 1.702f
#define LIMIT_ 7.0f
#define EPS_   1e-6f

// ---------------- WMMA fragment types ----------------
typedef __attribute__((ext_vector_type(16))) __bf16 v16bf;
typedef __attribute__((ext_vector_type(8)))  float  v8f;

union FragAB { v16bf v; unsigned int u[8]; };
union FragC  { v8f   v; float        f[8]; };

__device__ __forceinline__ unsigned short f2bf(float f) {
  unsigned int u = __float_as_uint(f);
  u = (u + 0x7FFFu + ((u >> 16) & 1u)) >> 16;
  return (unsigned short)u;
}
__device__ __forceinline__ float bf2f(unsigned short h) {
  return __uint_as_float(((unsigned int)h) << 16);
}

// Load one 16-bit A/B fragment (16x32 / 32x16) from a row-major row pointer.
// CDNA5 layout: lanes 0-15 -> K base 0, lanes 16-31 -> K base 8;
// VGPR v<4 holds K = base+2v..+1 ; v>=4 holds K = base+16+2(v-4)..+1
__device__ __forceinline__ void load_frag_pairs(FragAB& f, const unsigned short* rowptr, int lane) {
  const int kb = (lane & 16) ? 8 : 0;
#pragma unroll
  for (int v = 0; v < 4; ++v)
    f.u[v] = *(const unsigned int*)(rowptr + kb + 2 * v);
#pragma unroll
  for (int v = 0; v < 4; ++v)
    f.u[4 + v] = *(const unsigned int*)(rowptr + kb + 16 + 2 * v);
}

__device__ __forceinline__ void wmma_bf16(FragC& acc, const FragAB& a, const FragAB& b) {
  acc.v = __builtin_amdgcn_wmma_f32_16x16x32_bf16(false, a.v, false, b.v,
                                                  (short)0, acc.v, false, false);
}

__device__ __forceinline__ void zeroC(FragC& c) {
#pragma unroll
  for (int r = 0; r < 8; ++r) c.f[r] = 0.f;
}

// ---------------- RMSNorm: fp32 in -> bf16 out ----------------
__global__ __launch_bounds__(256) void rms_kernel(const float* __restrict__ X,
                                                  const float* __restrict__ w,
                                                  unsigned short* __restrict__ Y) {
  __shared__ float red[256];
  const int t = blockIdx.x;
  const float* x = X + (size_t)t * H_;
  float s = 0.f;
  for (int j = threadIdx.x; j < H_; j += 256) { float v = x[j]; s += v * v; }
  red[threadIdx.x] = s;
  __syncthreads();
  for (int off = 128; off > 0; off >>= 1) {
    if (threadIdx.x < off) red[threadIdx.x] += red[threadIdx.x + off];
    __syncthreads();
  }
  const float r = rsqrtf(red[0] / (float)H_ + EPS_);
  unsigned short* y = Y + (size_t)t * H_;
  for (int j = threadIdx.x; j < H_; j += 256) y[j] = f2bf(x[j] * r * w[j]);
}

// ---------------- generic WMMA GEMM (mode is compile-time) ----------------
// C[M,N] = A(bf16)[M,K] * W(f32->bf16)[K,N]  (+ bias / residual / GLU / scatter)
// Block tile 128x128, K-tile 64 (two WMMA k-steps per stage), 8 waves: each
// wave owns a 32x64 sub-tile (2 m-frags x 4 n-frags).
#define M_PLAIN  0
#define M_RESID  1
#define M_GATEUP 2
#define M_DOWN   3

template <int MODE>
__global__ __launch_bounds__(256) void gemm_kernel(
    const unsigned short* __restrict__ A, int ldA,
    const float* __restrict__ W, int N, int K,
    const float* __restrict__ bias,
    const float* __restrict__ resid, int ldR,
    float* __restrict__ Cout, int ldC,
    unsigned short* __restrict__ actOut, int ldAct,
    const int* __restrict__ rowList,
    const int* __restrict__ tokSlot,
    const int* __restrict__ countPtr,
    int M) {
  __shared__ unsigned short sA[128 * 64];   // [row][k]
  __shared__ unsigned short sB[128 * 64];   // [n][k] (transposed)

  const int tid  = threadIdx.x;
  const int lane = tid & 31;
  const int wave = tid >> 5;
  const int wm   = wave >> 1;   // 0..3
  const int wn   = wave & 1;    // 0..1
  const int bm   = blockIdx.y * 128;
  const int bn   = blockIdx.x * 128;  // physical column base

  const int Mact = countPtr ? *countPtr : M;
  if (bm >= Mact) return;

  // cooperative loader coordinates
  const int la_row = tid >> 1;          // 0..127
  const int la_col = (tid & 1) * 32;    // 0 / 32
  const int lb_k   = tid >> 2;          // 0..63
  const int lb_n   = (tid & 3) * 32;    // 0/32/64/96

  long long agrow = -1;
  {
    const int arow = bm + la_row;
    if (arow < Mact) agrow = rowList ? (long long)rowList[arow] : (long long)arow;
  }

  FragC acc[2][4];
#pragma unroll
  for (int i = 0; i < 2; ++i)
#pragma unroll
    for (int j = 0; j < 4; ++j) zeroC(acc[i][j]);

  const int nk = K >> 6;  // K-tiles of 64 (all K here are multiples of 64)
  for (int kk = 0; kk < nk; ++kk) {
    // ---- stage A tile (bf16), 128x64 ----
    {
      uint4 v0 = make_uint4(0, 0, 0, 0), v1 = v0, v2 = v0, v3 = v0;
      if (agrow >= 0) {
        const uint4* src = (const uint4*)(A + (size_t)agrow * ldA + kk * 64 + la_col);
        v0 = src[0]; v1 = src[1]; v2 = src[2]; v3 = src[3];
      }
      uint4* dst = (uint4*)(sA + la_row * 64 + la_col);
      dst[0] = v0; dst[1] = v1; dst[2] = v2; dst[3] = v3;
    }
    // ---- stage W tile (f32 -> bf16, transposed), 64x128 ----
    {
      const float* wrow = W + (size_t)(kk * 64 + lb_k) * N + bn + lb_n;
      if (bn + lb_n + 31 < N) {
        const float4* wsrc = (const float4*)wrow;
#pragma unroll
        for (int j4 = 0; j4 < 8; ++j4) {
          float4 f = wsrc[j4];
          sB[(lb_n + j4 * 4 + 0) * 64 + lb_k] = f2bf(f.x);
          sB[(lb_n + j4 * 4 + 1) * 64 + lb_k] = f2bf(f.y);
          sB[(lb_n + j4 * 4 + 2) * 64 + lb_k] = f2bf(f.z);
          sB[(lb_n + j4 * 4 + 3) * 64 + lb_k] = f2bf(f.w);
        }
      } else {
#pragma unroll
        for (int j = 0; j < 32; ++j) {
          const int n = bn + lb_n + j;
          sB[(lb_n + j) * 64 + lb_k] = f2bf((n < N) ? wrow[j] : 0.f);
        }
      }
      if (kk + 1 < nk)
        __builtin_prefetch(W + (size_t)((kk + 1) * 64 + lb_k) * N + bn + lb_n, 0, 1);
    }
    __syncthreads();

    // ---- WMMA: two k-steps of 32 over the staged 64 ----
#pragma unroll
    for (int ks = 0; ks < 2; ++ks) {
      FragAB af[2];
#pragma unroll
      for (int mt = 0; mt < 2; ++mt) {
        const int row = wm * 32 + mt * 16 + (lane & 15);
        load_frag_pairs(af[mt], sA + row * 64 + ks * 32, lane);
      }
      if (MODE == M_GATEUP) {
        // gate = even physical cols, up = odd -> separate accumulators, same lanes
#pragma unroll
        for (int nt = 0; nt < 2; ++nt)
#pragma unroll
          for (int p = 0; p < 2; ++p) {
            const int nloc = wn * 64 + nt * 32 + 2 * (lane & 15) + p;
            FragAB bf;
            load_frag_pairs(bf, sB + nloc * 64 + ks * 32, lane);
            wmma_bf16(acc[0][nt * 2 + p], af[0], bf);
            wmma_bf16(acc[1][nt * 2 + p], af[1], bf);
          }
      } else {
#pragma unroll
        for (int nt = 0; nt < 4; ++nt) {
          const int nloc = wn * 64 + nt * 16 + (lane & 15);
          FragAB bf;
          load_frag_pairs(bf, sB + nloc * 64 + ks * 32, lane);
          wmma_bf16(acc[0][nt], af[0], bf);
          wmma_bf16(acc[1][nt], af[1], bf);
        }
      }
    }
    __syncthreads();
  }

  // ---- epilogue ----
  const int hi = (lane >> 4) & 1;
  if (MODE == M_GATEUP) {
#pragma unroll
    for (int mt = 0; mt < 2; ++mt) {
      const int rowb = bm + wm * 32 + mt * 16;
#pragma unroll
      for (int nt = 0; nt < 2; ++nt) {
        const int clog = (bn >> 1) + wn * 32 + nt * 16 + (lane & 15);  // logical col in [0,F)
        const float bg = bias[2 * clog];
        const float bu = bias[2 * clog + 1];
#pragma unroll
        for (int r = 0; r < 8; ++r) {
          const int rr = rowb + r + hi * 8;
          if (rr < Mact) {
            float g = acc[mt][nt * 2 + 0].f[r] + bg;
            float u = acc[mt][nt * 2 + 1].f[r] + bu;
            g = fminf(g, LIMIT_);
            u = fminf(fmaxf(u, -LIMIT_), LIMIT_);
            const float glu = g / (1.f + __expf(-g * ALPHA_));
            actOut[(size_t)rr * ldAct + clog] = f2bf((u + 1.f) * glu);
          }
        }
      }
    }
  } else {
#pragma unroll
    for (int mt = 0; mt < 2; ++mt)
#pragma unroll
      for (int nt = 0; nt < 4; ++nt) {
        const int col = bn + wn * 64 + nt * 16 + (lane & 15);
        if (col < N) {
          const float bv = bias ? bias[col] : 0.f;
#pragma unroll
          for (int r = 0; r < 8; ++r) {
            const int rr = bm + wm * 32 + mt * 16 + r + hi * 8;
            if (rr < Mact) {
              float v = acc[mt][nt].f[r] + bv;
              if (MODE == M_RESID) v += resid[(size_t)rr * ldR + col];
              const size_t orow = (MODE == M_DOWN) ? (size_t)tokSlot[rr] : (size_t)rr;
              Cout[orow * ldC + col] = v;
            }
          }
        }
      }
  }
}

// ---------------- RoPE: qkv fp32 -> Q/K/V bf16 ----------------
__global__ __launch_bounds__(256) void rope_kernel(const float* __restrict__ qkv,
                                                   const float* __restrict__ cosb,
                                                   const float* __restrict__ sinb,
                                                   unsigned short* __restrict__ Qo,
                                                   unsigned short* __restrict__ Ko,
                                                   unsigned short* __restrict__ Vo) {
  const int t = blockIdx.x;
  const float* row = qkv + (size_t)t * QKV_N;
  const float* cr = cosb + (size_t)t * (HD_ / 2);
  const float* sr = sinb + (size_t)t * (HD_ / 2);
  for (int i = threadIdx.x; i < NH_ * 32; i += 256) {
    const int hh = i >> 5, d = i & 31;
    const float x1 = row[hh * 64 + d], x2 = row[hh * 64 + d + 32];
    const float c = cr[d], s = sr[d];
    Qo[(size_t)t * QDIM + hh * 64 + d]      = f2bf(x1 * c - x2 * s);
    Qo[(size_t)t * QDIM + hh * 64 + d + 32] = f2bf(x2 * c + x1 * s);
  }
  const float* kr = row + QDIM;
  for (int i = threadIdx.x; i < NKV_ * 32; i += 256) {
    const int hh = i >> 5, d = i & 31;
    const float x1 = kr[hh * 64 + d], x2 = kr[hh * 64 + d + 32];
    const float c = cr[d], s = sr[d];
    Ko[(size_t)t * KVDIM + hh * 64 + d]      = f2bf(x1 * c - x2 * s);
    Ko[(size_t)t * KVDIM + hh * 64 + d + 32] = f2bf(x2 * c + x1 * s);
  }
  const float* vr = row + QDIM + KVDIM;
  for (int i = threadIdx.x; i < KVDIM; i += 256)
    Vo[(size_t)t * KVDIM + i] = f2bf(vr[i]);
}

// ---------------- flash attention with sinks (WMMA) ----------------
__global__ __launch_bounds__(256) void attn_kernel(const unsigned short* __restrict__ Q,
                                                   const unsigned short* __restrict__ Kc,
                                                   const unsigned short* __restrict__ Vc,
                                                   const float* __restrict__ sinks,
                                                   unsigned short* __restrict__ O) {
  __shared__ unsigned short sK[64 * 64];      // [key][d]
  __shared__ unsigned short sV[64 * 64];      // [d][key]  (transposed)
  __shared__ float sS[8][16][64];             // per-wave score rows; repacked to bf16 in place
  __shared__ float sAl[8][16];

  const int tid  = threadIdx.x;
  const int lane = tid & 31;
  const int wave = tid >> 5;
  const int bh = blockIdx.y;
  const int b = bh / NH_;
  const int h = bh % NH_;
  const int kvh = h / (NH_ / NKV_);
  const int q0 = blockIdx.x * 128;
  const int qrow = q0 + wave * 16;
  const float scale = 0.125f;  // 1/sqrt(64)
  const int hi = (lane >> 4) & 1;

  FragAB qf[2];
  {
    const unsigned short* qbase = Q + (size_t)(b * S_ + qrow + (lane & 15)) * QDIM + h * HD_;
    load_frag_pairs(qf[0], qbase, lane);
    load_frag_pairs(qf[1], qbase + 32, lane);
  }

  float m = -1e30f, l = 0.f;
  FragC oacc[4];
#pragma unroll
  for (int nt = 0; nt < 4; ++nt) zeroC(oacc[nt]);

  const int kend = q0 + 128;  // keys < kend (causal)
  for (int k0 = 0; k0 < kend; k0 += 64) {
    // stage K (row major) and V (transposed)
    {
      const int key = tid >> 2;
      const int dseg = (tid & 3) * 16;
      const uint4* ksrc = (const uint4*)(Kc + (size_t)(b * S_ + k0 + key) * KVDIM + kvh * HD_ + dseg);
      uint4 v0 = ksrc[0], v1 = ksrc[1];
      uint4* kdst = (uint4*)(sK + key * 64 + dseg);
      kdst[0] = v0;
      kdst[1] = v1;
      const unsigned short* vsrc = Vc + (size_t)(b * S_ + k0 + key) * KVDIM + kvh * HD_ + dseg;
#pragma unroll
      for (int j = 0; j < 16; ++j) sV[(dseg + j) * 64 + key] = vsrc[j];
    }
    __syncthreads();

    // scores = Q K^T
    FragC sc[4];
#pragma unroll
    for (int nt = 0; nt < 4; ++nt) {
      zeroC(sc[nt]);
      const int keyl = nt * 16 + (lane & 15);
      FragAB bf;
      load_frag_pairs(bf, sK + keyl * 64, lane);
      wmma_bf16(sc[nt], qf[0], bf);
      load_frag_pairs(bf, sK + keyl * 64 + 32, lane);
      wmma_bf16(sc[nt], qf[1], bf);
    }
    // scaled + masked -> LDS
#pragma unroll
    for (int nt = 0; nt < 4; ++nt) {
      const int colg = k0 + nt * 16 + (lane & 15);
#pragma unroll
      for (int r = 0; r < 8; ++r) {
        const int rowg = qrow + r + hi * 8;
        float v = sc[nt].f[r] * scale;
        if (colg > rowg) v = -1e30f;
        sS[wave][r + hi * 8][nt * 16 + (lane & 15)] = v;
      }
    }
    __syncthreads();

    // online softmax per row (lanes 0..15 own one row each)
    if (lane < 16) {
      unsigned int* urow = (unsigned int*)(&sS[wave][lane][0]);
      float mt = -1e30f;
      for (int c = 0; c < 64; ++c) mt = fmaxf(mt, __uint_as_float(urow[c]));
      const float mn = fmaxf(m, mt);
      const float al = __expf(m - mn);
      float sum = 0.f;
      for (int j = 0; j < 32; ++j) {           // in-place repack float -> packed bf16 pair
        const float e0 = __expf(__uint_as_float(urow[2 * j]) - mn);
        const float e1 = __expf(__uint_as_float(urow[2 * j + 1]) - mn);
        sum += e0 + e1;
        urow[j] = (unsigned int)f2bf(e0) | ((unsigned int)f2bf(e1) << 16);
      }
      l = l * al + sum;
      m = mn;
      sAl[wave][lane] = al;
    }
    __syncthreads();

    // rescale running output, then O += P V
#pragma unroll
    for (int nt = 0; nt < 4; ++nt)
#pragma unroll
      for (int r = 0; r < 8; ++r) oacc[nt].f[r] *= sAl[wave][r + hi * 8];

    const unsigned short* prow = (const unsigned short*)(&sS[wave][lane & 15][0]);
#pragma unroll
    for (int c = 0; c < 2; ++c) {
      FragAB pf;
      load_frag_pairs(pf, prow + c * 32, lane);
#pragma unroll
      for (int nt = 0; nt < 4; ++nt) {
        const int d = nt * 16 + (lane & 15);
        FragAB vf;
        load_frag_pairs(vf, sV + d * 64 + c * 32, lane);
        wmma_bf16(oacc[nt], pf, vf);
      }
    }
    __syncthreads();
  }

  // fold in the attention sink, final normalize
  if (lane < 16) {
    const float s = sinks[h];
    const float mn = fmaxf(m, s);
    const float al = __expf(m - mn);
    const float li = l * al + __expf(s - mn);
    sAl[wave][lane] = al / li;
  }
  __syncthreads();

  const size_t orowbase = (size_t)(b * S_ + qrow);
#pragma unroll
  for (int nt = 0; nt < 4; ++nt) {
    const int col = h * HD_ + nt * 16 + (lane & 15);
#pragma unroll
    for (int r = 0; r < 8; ++r) {
      const float f = oacc[nt].f[r] * sAl[wave][r + hi * 8];
      O[(orowbase + r + hi * 8) * QDIM + col] = f2bf(f);
    }
  }
}

// ---------------- router: logits, softmax, top-2, token lists ----------------
__global__ __launch_bounds__(256) void router_kernel(const unsigned short* __restrict__ X,
                                                     const float* __restrict__ rw,
                                                     const float* __restrict__ rb,
                                                     int* __restrict__ counts,
                                                     int* __restrict__ list,
                                                     int* __restrict__ tokSlot,
                                                     float* __restrict__ wslot) {
  __shared__ float red[256 * 8];
  const int t = blockIdx.x;
  float p[8];
#pragma unroll
  for (int e = 0; e < 8; ++e) p[e] = 0.f;
  for (int j = threadIdx.x; j < H_; j += 256) {
    const float xv = bf2f(X[(size_t)t * H_ + j]);
    const float* r = rw + (size_t)j * E_;
#pragma unroll
    for (int e = 0; e < 8; ++e) p[e] += xv * r[e];
  }
#pragma unroll
  for (int e = 0; e < 8; ++e) red[threadIdx.x * 8 + e] = p[e];
  __syncthreads();
  for (int off = 128; off > 0; off >>= 1) {
    if (threadIdx.x < off)
#pragma unroll
      for (int e = 0; e < 8; ++e) red[threadIdx.x * 8 + e] += red[(threadIdx.x + off) * 8 + e];
    __syncthreads();
  }
  if (threadIdx.x == 0) {
    float lg[8];
    float mx = -1e30f;
#pragma unroll
    for (int e = 0; e < 8; ++e) { lg[e] = red[e] + rb[e]; mx = fmaxf(mx, lg[e]); }
    float den = 0.f;
#pragma unroll
    for (int e = 0; e < 8; ++e) { lg[e] = __expf(lg[e] - mx); den += lg[e]; }
    int e0 = 0;
#pragma unroll
    for (int e = 1; e < 8; ++e) if (lg[e] > lg[e0]) e0 = e;
    int e1 = (e0 == 0) ? 1 : 0;
#pragma unroll
    for (int e = 0; e < 8; ++e) if (e != e0 && lg[e] > lg[e1]) e1 = e;
    float w0 = lg[e0] / den, w1 = lg[e1] / den;
    const float ws = w0 + w1 + 1e-20f;
    w0 /= ws;
    w1 /= ws;
    const int p0 = atomicAdd(&counts[e0], 1);
    list[e0 * T_ + p0] = t;
    tokSlot[e0 * T_ + p0] = t * 2 + 0;
    wslot[t * 2 + 0] = w0;
    const int p1 = atomicAdd(&counts[e1], 1);
    list[e1 * T_ + p1] = t;
    tokSlot[e1 * T_ + p1] = t * 2 + 1;
    wslot[t * 2 + 1] = w1;
  }
}

// ---------------- final combine: residual + weighted expert slots ----------------
__global__ __launch_bounds__(256) void combine_kernel(const float* __restrict__ h2,
                                                      const float* __restrict__ oslot,
                                                      const float* __restrict__ wslot,
                                                      float* __restrict__ out) {
  const size_t i = (size_t)blockIdx.x * 256 + threadIdx.x;
  if (i >= (size_t)T_ * H_) return;
  const size_t t = i / H_;
  const size_t c = i % H_;
  out[i] = h2[i] + wslot[t * 2] * oslot[(t * 2) * H_ + c] +
           wslot[t * 2 + 1] * oslot[(t * 2 + 1) * H_ + c];
}

// ---------------- host launcher ----------------
extern "C" void kernel_launch(void* const* d_in, const int* in_sizes, int n_in,
                              void* d_out, int out_size, void* d_ws, size_t ws_size,
                              hipStream_t stream) {
  (void)in_sizes; (void)n_in; (void)out_size; (void)ws_size;

  const float* hidden = (const float*)d_in[0];
  const float* cosb   = (const float*)d_in[1];
  const float* sinb   = (const float*)d_in[2];
  const float* ln1w   = (const float*)d_in[3];
  const float* ln2w   = (const float*)d_in[4];
  const float* qkvw   = (const float*)d_in[5];
  const float* qkvb   = (const float*)d_in[6];
  const float* ow     = (const float*)d_in[7];
  const float* ob     = (const float*)d_in[8];
  const float* sinks  = (const float*)d_in[9];
  const float* rw     = (const float*)d_in[10];
  const float* rb     = (const float*)d_in[11];
  const float* gup    = (const float*)d_in[12];
  const float* gupb   = (const float*)d_in[13];
  const float* dwn    = (const float*)d_in[14];
  const float* dwnb   = (const float*)d_in[15];
  float* out = (float*)d_out;

  char* ws = (char*)d_ws;
  size_t off = 0;
  auto alloc = [&](size_t bytes) -> void* {
    void* p = ws + off;
    off = (off + bytes + 255) & ~(size_t)255;
    return p;
  };
  unsigned short* h1    = (unsigned short*)alloc((size_t)T_ * H_ * 2);
  float*          qkv   = (float*)alloc((size_t)T_ * QKV_N * 4);
  unsigned short* qb    = (unsigned short*)alloc((size_t)T_ * QDIM * 2);
  unsigned short* kb    = (unsigned short*)alloc((size_t)T_ * KVDIM * 2);
  unsigned short* vb    = (unsigned short*)alloc((size_t)T_ * KVDIM * 2);
  unsigned short* attnb = (unsigned short*)alloc((size_t)T_ * QDIM * 2);
  float*          h2    = (float*)alloc((size_t)T_ * H_ * 4);
  unsigned short* x2    = (unsigned short*)alloc((size_t)T_ * H_ * 2);
  int*            counts  = (int*)alloc(E_ * 4);
  int*            list    = (int*)alloc((size_t)E_ * T_ * 4);
  int*            tokslot = (int*)alloc((size_t)E_ * T_ * 4);
  float*          wslot   = (float*)alloc((size_t)T_ * 2 * 4);
  unsigned short* act     = (unsigned short*)alloc((size_t)T_ * F_ * 2);
  float*          oslot   = (float*)alloc((size_t)T_ * 2 * H_ * 4);

  // 1) RMSNorm 1
  rms_kernel<<<T_, 256, 0, stream>>>(hidden, ln1w, h1);

  // 2) QKV projection: [2048,2880] x [2880,5120]
  gemm_kernel<M_PLAIN><<<dim3(QKV_N / 128, T_ / 128), 256, 0, stream>>>(
      h1, H_, qkvw, QKV_N, H_, qkvb, nullptr, 0, qkv, QKV_N,
      nullptr, 0, nullptr, nullptr, nullptr, T_);

  // 3) RoPE + bf16 pack
  rope_kernel<<<T_, 256, 0, stream>>>(qkv, cosb, sinb, qb, kb, vb);

  // 4) flash attention with sinks
  attn_kernel<<<dim3(S_ / 128, B_ * NH_), 256, 0, stream>>>(qb, kb, vb, sinks, attnb);

  // 5) output projection + residual: [2048,4096] x [4096,2880]
  gemm_kernel<M_RESID><<<dim3((H_ + 127) / 128, T_ / 128), 256, 0, stream>>>(
      attnb, QDIM, ow, H_, QDIM, ob, hidden, H_, h2, H_,
      nullptr, 0, nullptr, nullptr, nullptr, T_);

  // 6) RMSNorm 2
  rms_kernel<<<T_, 256, 0, stream>>>(h2, ln2w, x2);

  // 7) router + top-2 token lists
  hipMemsetAsync(counts, 0, E_ * sizeof(int), stream);
  router_kernel<<<T_, 256, 0, stream>>>(x2, rw, rb, counts, list, tokslot, wslot);

  // 8) per-expert gate_up (fused clipped SwiGLU) and down (scatter to slots)
  for (int e = 0; e < E_; ++e) {
    gemm_kernel<M_GATEUP><<<dim3((2 * F_) / 128, T_ / 128), 256, 0, stream>>>(
        x2, H_, gup + (size_t)e * H_ * 2 * F_, 2 * F_, H_,
        gupb + (size_t)e * 2 * F_, nullptr, 0, nullptr, 0, act, F_,
        list + (size_t)e * T_, nullptr, counts + e, T_);
    gemm_kernel<M_DOWN><<<dim3((H_ + 127) / 128, T_ / 128), 256, 0, stream>>>(
        act, F_, dwn + (size_t)e * F_ * H_, H_, F_,
        dwnb + (size_t)e * H_, nullptr, 0, oslot, H_, nullptr, 0,
        nullptr, tokslot + (size_t)e * T_, counts + e, T_);
  }

  // 9) residual + weighted combine
  combine_kernel<<<(int)(((size_t)T_ * H_ + 255) / 256), 256, 0, stream>>>(h2, oslot, wslot, out);
}
